// GINE_69071664054687
// MI455X (gfx1250) — matrix operand order repaired and checked by reading, not compiled, not run on hardware
//
#include <hip/hip_runtime.h>
#include <hip/hip_bf16.h>

// ---------------------------------------------------------------------------
// GINE GNN forward for MI455X (gfx1250, wave32, WMMA).
// Big GEMMs ([100000,256]x[256,256]) use v_wmma_f32_16x16x32_f16 with f32
// accumulation; scatter/BN/pool/protein-conv are VALU + atomics + LDS.
// GEMM: one wave = 32x64 output strip (2 A frags, 8 accumulators, B reused).
// ---------------------------------------------------------------------------

#define NNODES   100000
#define NEDGES   300000
#define HID      256
#define NGRAPHS  256
#define SEQLEN   512
#define EMBD     128

typedef __attribute__((ext_vector_type(16))) _Float16 v16h;
typedef __attribute__((ext_vector_type(8)))  _Float16 v8h;
typedef __attribute__((ext_vector_type(4)))  _Float16 v4h;
typedef __attribute__((ext_vector_type(8)))  float    v8f;

// -------------------------------- utility ----------------------------------

__global__ void fill_f32(float* __restrict__ p, float v, size_t n) {
  size_t i = blockIdx.x * (size_t)blockDim.x + threadIdx.x;
  if (i < n) p[i] = v;
}

// transpose-convert weight [K=256, N=256] f32  ->  Wt [N][K] f16
__global__ __launch_bounds__(256)
void convert_wT(const float* __restrict__ W, _Float16* __restrict__ Wt) {
  int n = blockIdx.x;      // 0..255 (output column)
  int k = threadIdx.x;     // 0..255
  Wt[(size_t)n * HID + k] = (_Float16)W[(size_t)k * HID + n];
}

// repack protein conv weight [oc][ic][k] -> [oc][k][ic] (contiguous ic)
__global__ void repack_protw(const float* __restrict__ w, float* __restrict__ wr) {
  int t = blockIdx.x * blockDim.x + threadIdx.x;   // 256*128*5
  if (t >= 256 * EMBD * 5) return;
  int k  = t % 5;
  int ic = (t / 5) % EMBD;
  int oc = t / (5 * EMBD);
  wr[((size_t)oc * 5 + k) * EMBD + ic] = w[t];
}

// --------------------------- edge encoder (K=16) ---------------------------

__global__ __launch_bounds__(256)
void edge_enc(const float* __restrict__ attr,   // [E,16]
              const float* __restrict__ W,      // [16,256]
              const float* __restrict__ bias,   // [256]
              _Float16* __restrict__ ea, int E) {
  __shared__ float arow[16];
  int e = blockIdx.x;
  if (e >= E) return;
  int c = threadIdx.x;
  if (c < 16) arow[c] = attr[(size_t)e * 16 + c];
  __syncthreads();
  float acc = bias[c];
#pragma unroll
  for (int k = 0; k < 16; ++k) acc += arow[k] * W[k * HID + c];
  ea[(size_t)e * HID + c] = (_Float16)acc;
}

// ------------------------- message + scatter-add ---------------------------
// msg = relu(x[src] + ea); atomic scatter-add into aggr[dst]. 64 thr/edge x4 cols.

__global__ __launch_bounds__(256)
void scatter_msg(const int* __restrict__ src, const int* __restrict__ dst,
                 const float* __restrict__ x, const _Float16* __restrict__ ea,
                 float* __restrict__ aggr, int E) {
  long long t = blockIdx.x * (long long)blockDim.x + threadIdx.x;
  int e = (int)(t >> 6);
  if (e >= E) return;
  int c = ((int)t & 63) * 4;
  int s = src[e], d = dst[e];
  const float4 xv = *(const float4*)(x + (size_t)s * HID + c);
  const v4h ev = *(const v4h*)(ea + (size_t)e * HID + c);
  float* ap = aggr + (size_t)d * HID + c;
  atomicAdd(ap + 0, fmaxf(xv.x + (float)ev[0], 0.f));
  atomicAdd(ap + 1, fmaxf(xv.y + (float)ev[1], 0.f));
  atomicAdd(ap + 2, fmaxf(xv.z + (float)ev[2], 0.f));
  atomicAdd(ap + 3, fmaxf(xv.w + (float)ev[3], 0.f));
}

// z16 = f16(x + aggr)
__global__ void add_conv_h(const float* __restrict__ x, const float* __restrict__ aggr,
                           _Float16* __restrict__ out, size_t total) {
  size_t i = blockIdx.x * (size_t)blockDim.x + threadIdx.x;
  if (i < total) out[i] = (_Float16)(x[i] + aggr[i]);
}

// ------------------------------ WMMA GEMM ----------------------------------
// C[M,256] = A16[M,256] @ W (given as Wt[N=256][K=256] f16) + bias.
// One wave -> 32x64 strip: 2 A fragments, 8 f32 accumulators, each B fragment
// reused for both row tiles. 4 waves/block -> 32x256 row band. M % 32 == 0.

__global__ __launch_bounds__(128)
void gemm_wmma_256(const _Float16* __restrict__ A,
                   const _Float16* __restrict__ Wt,
                   const float* __restrict__ bias,
                   float* __restrict__ C, int M) {
  const int lane = threadIdx.x & 31;
  const int wave = threadIdx.x >> 5;        // 0..3
  const int m0   = blockIdx.x * 32;
  if (m0 >= M) return;
  const int n0    = wave * 64;
  const int koff  = (lane >> 4) * 8;        // A fragment half-group
  const int kboff = (lane >> 4) * 16;       // B fragment half-group
  const int nlane = lane & 15;

  v8f acc[2][4] = {};
  const _Float16* arow0 = A + (size_t)(m0 + (lane & 15)) * HID;
  const _Float16* arow1 = arow0 + (size_t)16 * HID;

  for (int kc = 0; kc < HID; kc += 32) {
    union { v16h v; v8h h[2]; } a0, a1;
    a0.h[0] = *(const v8h*)(arow0 + kc + koff);        // K = kc+koff   .. +7
    a0.h[1] = *(const v8h*)(arow0 + kc + 16 + koff);   // K = kc+16+off .. +7
    a1.h[0] = *(const v8h*)(arow1 + kc + koff);
    a1.h[1] = *(const v8h*)(arow1 + kc + 16 + koff);
#pragma unroll
    for (int j = 0; j < 4; ++j) {
      const _Float16* bp = Wt + (size_t)(n0 + j * 16 + nlane) * HID + kc + (kboff >> 1);
      union { v16h v; v8h h[2]; } bu;
      bu.h[0] = *(const v8h*)bp;                       // K = kc+8h .. +7
      bu.h[1] = *(const v8h*)(bp + 16);                // K = kc+16+8h .. +7
      v16h b = bu.v;
      acc[0][j] = __builtin_amdgcn_wmma_f32_16x16x32_f16(
          false, a0.v, false, b, (short)0, acc[0][j], false, false);
      acc[1][j] = __builtin_amdgcn_wmma_f32_16x16x32_f16(
          false, a1.v, false, b, (short)0, acc[1][j], false, false);
    }
  }

#pragma unroll
  for (int mi = 0; mi < 2; ++mi) {
#pragma unroll
    for (int j = 0; j < 4; ++j) {
      int n = n0 + j * 16 + nlane;
      float bv = bias[n];
#pragma unroll
      for (int r = 0; r < 8; ++r) {
        int m = m0 + mi * 16 + r + ((lane >> 4) << 3);
        C[(size_t)m * HID + n] = acc[mi][j][r] + bv;
      }
    }
  }
}

// ------------------------------ batchnorm ----------------------------------

__global__ __launch_bounds__(256)
void bn_partial(const float* __restrict__ X, float* __restrict__ sum,
                float* __restrict__ sumsq, int M, int cols, int rowsPerBlock) {
  int c = threadIdx.x;
  if (c >= cols) return;
  int r0 = blockIdx.x * rowsPerBlock;
  int r1 = min(r0 + rowsPerBlock, M);
  float s = 0.f, q = 0.f;
  for (int r = r0; r < r1; ++r) {
    float v = X[(size_t)r * cols + c];
    s += v; q += v * v;
  }
  atomicAdd(&sum[c], s);
  atomicAdd(&sumsq[c], q);
}

__global__ void bn_finalize(const float* __restrict__ sum, const float* __restrict__ sumsq,
                            const float* __restrict__ g, const float* __restrict__ b,
                            float* __restrict__ scale, float* __restrict__ shift,
                            int M, int cols) {
  int c = blockIdx.x * blockDim.x + threadIdx.x;
  if (c >= cols) return;
  float inv = 1.0f / (float)M;
  float m = sum[c] * inv;
  float v = sumsq[c] * inv - m * m;
  float rs = rsqrtf(v + 1e-5f);
  float sc = g[c] * rs;
  scale[c] = sc;
  shift[c] = b[c] - m * sc;
}

// out_h = f16(relu(X*scale + shift))
__global__ void bn_relu_h(const float* __restrict__ X, const float* __restrict__ scale,
                          const float* __restrict__ shift, _Float16* __restrict__ out,
                          size_t total, int cmask) {
  size_t i = blockIdx.x * (size_t)blockDim.x + threadIdx.x;
  if (i >= total) return;
  int c = (int)(i & (size_t)cmask);
  out[i] = (_Float16)fmaxf(X[i] * scale[c] + shift[c], 0.f);
}

// x = relu([x +] y*scale + shift)
__global__ void bn_combine(const float* __restrict__ y, const float* __restrict__ scale,
                           const float* __restrict__ shift, float* __restrict__ x,
                           size_t total, int addRes) {
  size_t i = blockIdx.x * (size_t)blockDim.x + threadIdx.x;
  if (i >= total) return;
  int c = (int)(i & 255);
  float v = y[i] * scale[c] + shift[c];
  if (addRes) v += x[i];
  x[i] = fmaxf(v, 0.f);
}

// X = relu(X*scale + shift)  (small, f32, in place)
__global__ void bn_relu_f32(float* __restrict__ X, const float* __restrict__ scale,
                            const float* __restrict__ shift, size_t total, int cmask) {
  size_t i = blockIdx.x * (size_t)blockDim.x + threadIdx.x;
  if (i >= total) return;
  int c = (int)(i & (size_t)cmask);
  X[i] = fmaxf(X[i] * scale[c] + shift[c], 0.f);
}

// ------------------------------ mean pool ----------------------------------

__global__ __launch_bounds__(256)
void pool_accum(const float* __restrict__ x, const int* __restrict__ batch,
                float* __restrict__ gsum, float* __restrict__ cnt, int Nn) {
  int n = blockIdx.x;
  if (n >= Nn) return;
  int c = threadIdx.x;
  int g = batch[n];
  atomicAdd(&gsum[(size_t)g * HID + c], x[(size_t)n * HID + c]);
  if (c == 0) atomicAdd(&cnt[g], 1.0f);
}

__global__ __launch_bounds__(256)
void pool_div(const float* __restrict__ gsum, const float* __restrict__ cnt,
              float* __restrict__ feat) {
  int t = blockIdx.x * blockDim.x + threadIdx.x;    // 0..65535
  int g = t >> 8, c = t & 255;
  feat[(size_t)g * 512 + c] = gsum[t] / fmaxf(cnt[g], 1.0f);
}

// ------------------- protein: embed -> conv1d(k5,p2) -> relu -> maxpool ----
// block = one graph; thread = one output channel; seq staged in LDS chunks.
// weights pre-repacked to [oc][k][ic] so inner ic loop is contiguous.

__global__ __launch_bounds__(256)
void prot_conv(const int* __restrict__ seq, const float* __restrict__ emb,
               const float* __restrict__ wr,     // [256][5][128]
               const float* __restrict__ bias,   // [256]
               float* __restrict__ feat) {       // [256][512], write cols 256..511
  __shared__ float sE[100 * EMBD];               // 96-chunk + 4 halo = 50 KB
  int b  = blockIdx.x;
  int oc = threadIdx.x;
  const float* wrow = wr + (size_t)oc * 5 * EMBD;
  float best = 0.f;                               // max over relu >= 0

  for (int s0 = 0; s0 < SEQLEN; s0 += 96) {
    int lo = s0 - 2;
    for (int idx = threadIdx.x; idx < 100 * EMBD; idx += blockDim.x) {
      int p  = lo + idx / EMBD;
      int ic = idx % EMBD;
      float v = 0.f;
      if (p >= 0 && p < SEQLEN)
        v = emb[(size_t)seq[(size_t)b * SEQLEN + p] * EMBD + ic];
      sE[idx] = v;
    }
    __syncthreads();
    int smax = min(96, SEQLEN - s0);
    for (int si = 0; si < smax; ++si) {
      float acc = bias[oc];
#pragma unroll
      for (int k = 0; k < 5; ++k) {
        const float* ev = sE + (si + k) * EMBD;   // pos = s0+si-2+k
        const float* wv = wrow + k * EMBD;        // contiguous over ic
        for (int ic = 0; ic < EMBD; ++ic)
          acc = fmaf(ev[ic], wv[ic], acc);
      }
      best = fmaxf(best, acc);
    }
    __syncthreads();
  }
  feat[(size_t)b * 512 + 256 + oc] = fmaxf(best, 0.f);
}

// ------------------------ small dense FC (final MLP) -----------------------

__global__ void fc_small(const float* __restrict__ A, const float* __restrict__ W,
                         const float* __restrict__ b, float* __restrict__ C,
                         int M, int K, int N) {
  int t = blockIdx.x * blockDim.x + threadIdx.x;
  if (t >= M * N) return;
  int m = t / N, n = t % N;
  float acc = b[n];
  for (int k = 0; k < K; ++k) acc = fmaf(A[(size_t)m * K + k], W[(size_t)k * N + n], acc);
  C[t] = acc;
}

// ---------------------------------------------------------------------------

extern "C" void kernel_launch(void* const* d_in, const int* in_sizes, int n_in,
                              void* d_out, int out_size, void* d_ws, size_t ws_size,
                              hipStream_t stream) {
  (void)n_in; (void)out_size; (void)ws_size;
  // ---- input indices (setup_inputs dict insertion order, recursively) ----
  // 0 h[100000,256]  1 edge_index[2,E](int)  2 edge_attr[E,16]  3 batch(int)
  // 4 prot_seq(int)  5 edge_enc.w[16,256]    6 edge_enc.b[256]
  // conv i (i=0..4) base 7+10i: l1.w,l1.b,bn1.g,bn1.b,l2.w,l2.b,bn2.g,bn2.b,l3.w,l3.b
  // norms i: g=57+2i, b=58+2i
  // 67 prot.emb[26,128]  68 prot.conv_w[256,128,5]  69 prot.conv_b[256]
  // final base 70: l1.w[512,128],l1.b,bn1.g,bn1.b,l2.w[128,128],l2.b,bn2.g,bn2.b,l3.w[128,1],l3.b[1]
  auto F = [&](int i) { return (const float*)d_in[i]; };
  auto I = [&](int i) { return (const int*)d_in[i]; };

  const int Nn = in_sizes[0] / HID;      // 100000
  const int E  = in_sizes[1] / 2;        // 300000
  const float* h     = F(0);
  const int*   srcI  = I(1);
  const int*   dstI  = I(1) + E;
  const int*   batch = I(3);
  const int*   pseq  = I(4);

  // ---- workspace carve-out (256-B aligned) ----
  size_t off = 0;
  auto carve = [&](size_t bytes) {
    void* p = (char*)d_ws + off;
    off += (bytes + 255) & ~(size_t)255;
    return p;
  };
  _Float16* ea   = (_Float16*)carve((size_t)E  * HID * 2);
  float*    x    = (float*)   carve((size_t)Nn * HID * 4);
  float*    aggr = (float*)   carve((size_t)Nn * HID * 4);
  float*    t1   = (float*)   carve((size_t)Nn * HID * 4);   // also GEMM3 output y
  float*    t2   = (float*)   carve((size_t)Nn * HID * 4);
  _Float16* ah   = (_Float16*)carve((size_t)Nn * HID * 2);   // f16 GEMM input
  _Float16* w16t = (_Float16*)carve((size_t)HID * HID * 2);
  float*    sums = (float*)   carve(4 * 256 * 4);            // sums|sumsq|scale|shift
  float*    sumsq = sums + 256, *scale = sums + 512, *shift = sums + 768;
  float*    gsum = (float*)   carve((size_t)NGRAPHS * HID * 4);
  float*    cnt  = (float*)   carve((size_t)NGRAPHS * 4);
  float*    feat = (float*)   carve((size_t)NGRAPHS * 512 * 4);
  float*    f1   = (float*)   carve((size_t)NGRAPHS * 128 * 4);
  float*    f2   = (float*)   carve((size_t)NGRAPHS * 128 * 4);
  float*    wrp  = (float*)   carve((size_t)256 * EMBD * 5 * 4);  // repacked prot weights

  const size_t tot = (size_t)Nn * HID;
  const unsigned gElem = (unsigned)((tot + 255) / 256);

  auto run_bn = [&](float* X, int M, int cols, const float* g, const float* b) {
    fill_f32<<<2, 256, 0, stream>>>(sums, 0.f, 512);
    int rpb = (cols == HID) ? 512 : M;
    unsigned gb = (unsigned)((M + rpb - 1) / rpb);
    bn_partial<<<gb, 256, 0, stream>>>(X, sums, sumsq, M, cols, rpb);
    bn_finalize<<<1, 256, 0, stream>>>(sums, sumsq, g, b, scale, shift, M, cols);
  };

  // ---- edge encoder: ea = f16(edge_attr @ W + b) ----
  edge_enc<<<E, 256, 0, stream>>>(F(2), F(5), F(6), ea, E);

  // ---- 5 GINE conv layers ----
  for (int l = 0; l < 5; ++l) {
    const int base = 7 + 10 * l;
    const float* xin = (l == 0) ? h : x;

    fill_f32<<<gElem, 256, 0, stream>>>(aggr, 0.f, tot);
    scatter_msg<<<(unsigned)(((long long)E * 64 + 255) / 256), 256, 0, stream>>>(
        srcI, dstI, xin, ea, aggr, E);
    add_conv_h<<<gElem, 256, 0, stream>>>(xin, aggr, ah, tot);

    // GEMM1 + BN1 + relu
    convert_wT<<<256, 256, 0, stream>>>(F(base + 0), w16t);
    gemm_wmma_256<<<(unsigned)(Nn / 32), 128, 0, stream>>>(ah, w16t, F(base + 1), t1, Nn);
    run_bn(t1, Nn, HID, F(base + 2), F(base + 3));
    bn_relu_h<<<gElem, 256, 0, stream>>>(t1, scale, shift, ah, tot, 255);

    // GEMM2 + BN2 + relu
    convert_wT<<<256, 256, 0, stream>>>(F(base + 4), w16t);
    gemm_wmma_256<<<(unsigned)(Nn / 32), 128, 0, stream>>>(ah, w16t, F(base + 5), t2, Nn);
    run_bn(t2, Nn, HID, F(base + 6), F(base + 7));
    bn_relu_h<<<gElem, 256, 0, stream>>>(t2, scale, shift, ah, tot, 255);

    // GEMM3 -> y (in t1)
    convert_wT<<<256, 256, 0, stream>>>(F(base + 8), w16t);
    gemm_wmma_256<<<(unsigned)(Nn / 32), 128, 0, stream>>>(ah, w16t, F(base + 9), t1, Nn);

    // outer BN (norms[l]) + relu (+ residual for l>0) -> x
    run_bn(t1, Nn, HID, F(57 + 2 * l), F(58 + 2 * l));
    bn_combine<<<gElem, 256, 0, stream>>>(t1, scale, shift, x, tot, (l > 0) ? 1 : 0);
  }

  // ---- global mean pool -> feat[:, 0:256] ----
  fill_f32<<<(NGRAPHS * HID + 255) / 256, 256, 0, stream>>>(gsum, 0.f, (size_t)NGRAPHS * HID);
  fill_f32<<<1, 256, 0, stream>>>(cnt, 0.f, NGRAPHS);
  pool_accum<<<(unsigned)Nn, 256, 0, stream>>>(x, batch, gsum, cnt, Nn);
  pool_div<<<NGRAPHS, 256, 0, stream>>>(gsum, cnt, feat);

  // ---- protein branch -> feat[:, 256:512] ----
  repack_protw<<<(256 * EMBD * 5 + 255) / 256, 256, 0, stream>>>(F(68), wrp);
  prot_conv<<<NGRAPHS, 256, 0, stream>>>(pseq, F(67), wrp, F(69), feat);

  // ---- final MLP: [256,512] -> 128 (BN,relu) -> 128 (BN,relu) -> 1 ----
  fc_small<<<(NGRAPHS * 128 + 255) / 256, 256, 0, stream>>>(feat, F(70), F(71), f1,
                                                            NGRAPHS, 512, 128);
  run_bn(f1, NGRAPHS, 128, F(72), F(73));
  bn_relu_f32<<<(NGRAPHS * 128 + 255) / 256, 256, 0, stream>>>(f1, scale, shift,
                                                               (size_t)NGRAPHS * 128, 127);
  fc_small<<<(NGRAPHS * 128 + 255) / 256, 256, 0, stream>>>(f1, F(74), F(75), f2,
                                                            NGRAPHS, 128, 128);
  run_bn(f2, NGRAPHS, 128, F(76), F(77));
  bn_relu_f32<<<(NGRAPHS * 128 + 255) / 256, 256, 0, stream>>>(f2, scale, shift,
                                                               (size_t)NGRAPHS * 128, 127);
  fc_small<<<1, 256, 0, stream>>>(f2, F(78), F(79), (float*)d_out, NGRAPHS, 128, 1);
}
